// GRU_GNN_Model_targettest_52467320488320
// MI455X (gfx1250) — compile-verified, matrix operand order checked by reading, not run on hardware
//
#include <hip/hip_runtime.h>
#include <hip/hip_bf16.h>
#include <math.h>

// ---------------------------------------------------------------------------
// GNN VAE decoder for MI455X (gfx1250).
// Heavy path: EdgeConv2/3 message GEMMs via v_wmma_f32_16x16x32_bf16.
// EdgeConv1 collapses analytically (z is broadcast across nodes).
// Segment-max finalize (-inf -> 0) is fused into the next consumer's reads.
// ---------------------------------------------------------------------------

typedef __attribute__((ext_vector_type(16))) __bf16 v16bf;
typedef __attribute__((ext_vector_type(8)))  float  v8f;

#define H256 256
#define DIN  1024
#define EPB  32     // edges per WMMA block (2 M-tiles)

__device__ __forceinline__ unsigned short f32_to_bf16(float f) {
  unsigned u = __float_as_uint(f);
  unsigned r = u + 0x7FFFu + ((u >> 16) & 1u);   // round-to-nearest-even
  return (unsigned short)(r >> 16);
}

// float atomic-max via integer atomics (exact IEEE ordering, no NaNs here).
__device__ __forceinline__ void atomicMaxF(float* addr, float val) {
  if (val >= 0.0f) atomicMax((int*)addr, __float_as_int(val));
  else             atomicMin((unsigned int*)addr, __float_as_uint(val));
}

// Empty-segment fixup fused at read time: -inf -> 0 (matches isfinite->0).
__device__ __forceinline__ float fixneg(float v) {
  return (v > -INFINITY) ? v : 0.0f;
}

// K-within-32 position held by (laneHi, halfIdx t) in a 16-bit WMMA fragment.
__device__ __forceinline__ int klocal(int hi, int t) {
  int base = (t < 8) ? t : (t + 8);   // t0..7 -> K0..7, t8..15 -> K16..23
  return base + hi * 8;               // high half-wave shifted by 8
}

// ---------------------------------------------------------------------------
// z = (x@Wzm+bzm) + exp((x@Wzlv+bzlv)/2) * eps          [B,1024]
// ---------------------------------------------------------------------------
__global__ __launch_bounds__(256) void k_sample_z(
    const float* __restrict__ x, const float* __restrict__ eps,
    const float* __restrict__ Wzm, const float* __restrict__ bzm,
    const float* __restrict__ Wzlv, const float* __restrict__ bzlv,
    float* __restrict__ z, int B) {
  int idx = blockIdx.x * 256 + threadIdx.x;
  if (idx >= B * DIN) return;
  int b = idx / DIN, c = idx % DIN;
  const float* xb = x + (size_t)b * DIN;
  float zm = bzm[c], lv = bzlv[c];
  for (int k = 0; k < DIN; ++k) {
    float xv = xb[k];
    zm += xv * Wzm[(size_t)k * DIN + c];
    lv += xv * Wzlv[(size_t)k * DIN + c];
  }
  z[idx] = zm + __expf(0.5f * lv) * eps[idx];
}

// ---------------------------------------------------------------------------
// zW1[b,c] = sum_k z[b,k] * W1[k,c] + b1[c]   (only rows 0..1023 of W1)
// ---------------------------------------------------------------------------
__global__ __launch_bounds__(256) void k_zw1(
    const float* __restrict__ z, const float* __restrict__ W1,
    const float* __restrict__ b1, float* __restrict__ zW1) {
  int b = blockIdx.x, c = threadIdx.x;
  const float* zb = z + (size_t)b * DIN;
  float acc = b1[c];
  for (int k = 0; k < DIN; ++k) acc += zb[k] * W1[(size_t)k * H256 + c];
  zW1[b * H256 + c] = acc;
}

// ---------------------------------------------------------------------------
// Pack a (512 x 256) f32 weight matrix into per-lane bf16 WMMA B fragments:
// out[((kt*16 + nt)*32 + lane)*16 + t] = bf16(W[(kt*32 + klocal)*256 + nt*16 + lane%16])
// ---------------------------------------------------------------------------
__global__ __launch_bounds__(256) void k_swizzle_w(
    const float* __restrict__ W, unsigned short* __restrict__ out, int K) {
  int idx = blockIdx.x * 256 + threadIdx.x;
  if (idx >= K * H256) return;
  int t    = idx & 15;
  int lane = (idx >> 4) & 31;
  int nt   = (idx >> 9) & 15;
  int kt   = idx >> 13;
  int k = kt * 32 + klocal(lane >> 4, t);
  int n = nt * 16 + (lane & 15);
  out[idx] = f32_to_bf16(W[(size_t)k * H256 + n]);
}

// ---------------------------------------------------------------------------
// Fill a float buffer with -inf (segment-max identity).
// ---------------------------------------------------------------------------
__global__ __launch_bounds__(256) void k_fill_neginf(float* __restrict__ p, size_t n) {
  size_t i = (size_t)blockIdx.x * 256 + threadIdx.x;
  if (i < n) p[i] = -INFINITY;
}

// ---------------------------------------------------------------------------
// EdgeConv-1 using the broadcast-z decomposition:
// m[c] = zW1[b,c] + pos_i . W1[1024:1027, c] + (pos_j - pos_i) . W1[2051:2054, c]
// ---------------------------------------------------------------------------
__global__ __launch_bounds__(256) void k_edgeconv1(
    const float* __restrict__ svp,      // [B, N*3]
    const int* __restrict__ esrc, const int* __restrict__ edst,
    const float* __restrict__ W1, const float* __restrict__ zW1,
    const float* __restrict__ a1, float* __restrict__ hout, int N) {
  int e = blockIdx.x, b = blockIdx.y, c = threadIdx.x;
  int si = esrc[e], di = edst[e];
  const float* pb = svp + (size_t)b * N * 3;
  float m = zW1[b * H256 + c];
  #pragma unroll
  for (int d = 0; d < 3; ++d) {
    float pi = pb[di * 3 + d];
    float pj = pb[si * 3 + d];
    m += pi * W1[(size_t)(1024 + d) * H256 + c];
    m += (pj - pi) * W1[(size_t)(2051 + d) * H256 + c];
  }
  float al = a1[c];
  m = (m >= 0.0f) ? m : al * m;
  atomicMaxF(&hout[((size_t)b * N + di) * H256 + c], m);
}

// ---------------------------------------------------------------------------
// EdgeConv-2/3 message GEMM + PReLU + segment-max.
// Block: 256 thr (8 waves) = 32 edges x 256 outputs.  K = 512, 16 WMMA k-steps.
// A tile (32 x 512 bf16, rows = edges, cols = [h[dst], h[src]-h[dst]]) in LDS;
// -inf fixup from the previous layer applied on read. Each wave owns 2 N-tiles
// across 2 M-tiles (4 accumulators); B fragments pre-swizzled bf16 in global,
// each load amortized over both M-tiles.
// ---------------------------------------------------------------------------
#define LDA 520   // halves per LDS row (pad: 1040B -> rows offset 4 banks apart)

union FragBF { v16bf v; uint4 q[2]; };

__global__ __launch_bounds__(256) void k_edgeconv_wmma(
    const float* __restrict__ hprev,          // [B, N, 256] (raw, may hold -inf)
    const int* __restrict__ esrc, const int* __restrict__ edst,
    const unsigned short* __restrict__ Wsw,   // swizzled (512x256) bf16
    const float* __restrict__ bias, const float* __restrict__ alpha,
    float* __restrict__ hout, int N) {
  int b  = blockIdx.y;
  int e0 = blockIdx.x * EPB;
  const float* hp = hprev + (size_t)b * N * H256;
  float*       ho = hout  + (size_t)b * N * H256;

  __shared__ __align__(16) unsigned short ldsA[EPB * LDA];
  __shared__ int sdst[EPB], ssrc[EPB];

  int tid = threadIdx.x;
  if (tid < EPB)           sdst[tid] = edst[e0 + tid];
  else if (tid < 2 * EPB)  ssrc[tid - EPB] = esrc[e0 + tid - EPB];
  __syncthreads();

  // Cooperative gather: one (edge, k<256) pair per element -> both LDS halves.
  #pragma unroll
  for (int it = 0; it < EPB; ++it) {           // 32*256 / 256 = 32
    int i  = tid + it * 256;
    int er = i >> 8;
    int kk = i & 255;
    float vi = fixneg(hp[(size_t)sdst[er] * H256 + kk]);
    float vj = fixneg(hp[(size_t)ssrc[er] * H256 + kk]);
    ldsA[er * LDA + kk]       = f32_to_bf16(vi);        // xi
    ldsA[er * LDA + 256 + kk] = f32_to_bf16(vj - vi);   // xj - xi
  }
  __syncthreads();

  int wave = tid >> 5;
  int lane = tid & 31;
  int row  = lane & 15;   // edge row (M%16) for A; N%16 for B/C/D
  int hi   = lane >> 4;

  int nt0 = wave * 2, nt1 = wave * 2 + 1;
  v8f acc00 = {}; v8f acc01 = {};   // M-tile 0 x {nt0, nt1}
  v8f acc10 = {}; v8f acc11 = {};   // M-tile 1 x {nt0, nt1}

  #pragma unroll
  for (int kt = 0; kt < 16; ++kt) {
    FragBF fa0, fa1;
    const unsigned short* ap0 = &ldsA[row * LDA + kt * 32 + hi * 8];
    const unsigned short* ap1 = ap0 + 16 * LDA;
    fa0.q[0] = *(const uint4*)(ap0); fa0.q[1] = *(const uint4*)(ap0 + 16);
    fa1.q[0] = *(const uint4*)(ap1); fa1.q[1] = *(const uint4*)(ap1 + 16);

    FragBF fb0, fb1;
    const uint4* bp0 = (const uint4*)(Wsw + ((size_t)(kt * 16 + nt0) * 32 + lane) * 16);
    const uint4* bp1 = (const uint4*)(Wsw + ((size_t)(kt * 16 + nt1) * 32 + lane) * 16);
    fb0.q[0] = bp0[0]; fb0.q[1] = bp0[1];
    fb1.q[0] = bp1[0]; fb1.q[1] = bp1[1];

    acc00 = __builtin_amdgcn_wmma_f32_16x16x32_bf16(false, fa0.v, false, fb0.v,
                                                    (short)0, acc00, false, false);
    acc01 = __builtin_amdgcn_wmma_f32_16x16x32_bf16(false, fa0.v, false, fb1.v,
                                                    (short)0, acc01, false, false);
    acc10 = __builtin_amdgcn_wmma_f32_16x16x32_bf16(false, fa1.v, false, fb0.v,
                                                    (short)0, acc10, false, false);
    acc11 = __builtin_amdgcn_wmma_f32_16x16x32_bf16(false, fa1.v, false, fb1.v,
                                                    (short)0, acc11, false, false);
  }

  // Epilogue: bias + PReLU + atomic segment-max at dst.
  int col0 = nt0 * 16 + row;
  int col1 = nt1 * 16 + row;
  float bi0 = bias[col0], al0 = alpha[col0];
  float bi1 = bias[col1], al1 = alpha[col1];
  #pragma unroll
  for (int r = 0; r < 8; ++r) {
    int m0 = r + hi * 8;            // edge row per C/D layout, M-tile 0
    int m1 = m0 + 16;               // M-tile 1
    size_t d0 = (size_t)sdst[m0] * H256;
    size_t d1 = (size_t)sdst[m1] * H256;
    float v;
    v = acc00[r] + bi0; v = (v >= 0.0f) ? v : al0 * v; atomicMaxF(&ho[d0 + col0], v);
    v = acc01[r] + bi1; v = (v >= 0.0f) ? v : al1 * v; atomicMaxF(&ho[d0 + col1], v);
    v = acc10[r] + bi0; v = (v >= 0.0f) ? v : al0 * v; atomicMaxF(&ho[d1 + col0], v);
    v = acc11[r] + bi1; v = (v >= 0.0f) ? v : al1 * v; atomicMaxF(&ho[d1 + col1], v);
  }
}

// ---------------------------------------------------------------------------
// y[b,n,:] = fixneg(h3[b,n,:]) @ Wo(256x3) + bo
// ---------------------------------------------------------------------------
__global__ __launch_bounds__(256) void k_outproj(
    const float* __restrict__ h3, const float* __restrict__ Wo,
    const float* __restrict__ bo, float* __restrict__ y, int BN) {
  int idx = blockIdx.x * 256 + threadIdx.x;
  if (idx >= BN) return;
  const float* h = h3 + (size_t)idx * H256;
  float a0 = bo[0], a1 = bo[1], a2 = bo[2];
  for (int k = 0; k < H256; ++k) {
    float hv = fixneg(h[k]);
    a0 += hv * Wo[k * 3 + 0];
    a1 += hv * Wo[k * 3 + 1];
    a2 += hv * Wo[k * 3 + 2];
  }
  y[(size_t)idx * 3 + 0] = a0;
  y[(size_t)idx * 3 + 1] = a1;
  y[(size_t)idx * 3 + 2] = a2;
}

// ---------------------------------------------------------------------------
extern "C" void kernel_launch(void* const* d_in, const int* in_sizes, int n_in,
                              void* d_out, int out_size, void* d_ws, size_t ws_size,
                              hipStream_t stream) {
  const float* x    = (const float*)d_in[0];
  const float* svp  = (const float*)d_in[1];
  const float* eps  = (const float*)d_in[2];
  const int*   eidx = (const int*)  d_in[3];
  const float* Wzm  = (const float*)d_in[4];
  const float* bzm  = (const float*)d_in[5];
  const float* Wzlv = (const float*)d_in[6];
  const float* bzlv = (const float*)d_in[7];
  const float* W1   = (const float*)d_in[8];
  const float* b1   = (const float*)d_in[9];
  const float* a1   = (const float*)d_in[10];
  const float* W2   = (const float*)d_in[11];
  const float* b2   = (const float*)d_in[12];
  const float* a2   = (const float*)d_in[13];
  const float* W3   = (const float*)d_in[14];
  const float* b3   = (const float*)d_in[15];
  const float* a3   = (const float*)d_in[16];
  const float* Wo   = (const float*)d_in[17];
  const float* bo   = (const float*)d_in[18];
  float* y = (float*)d_out;

  const int B = in_sizes[0] / DIN;
  const int N = in_sizes[1] / (3 * B);
  const int E = in_sizes[3] / 2;
  const int* esrc = eidx;
  const int* edst = eidx + E;

  // Workspace layout
  char* ws = (char*)d_ws;
  size_t hbytes = (size_t)B * N * H256 * sizeof(float);   // 32 MB
  float* buf0 = (float*)ws;                 ws += hbytes;
  float* buf1 = (float*)ws;                 ws += hbytes;
  float* z    = (float*)ws;                 ws += (size_t)B * DIN * sizeof(float);
  float* zW1  = (float*)ws;                 ws += (size_t)B * H256 * sizeof(float);
  unsigned short* W2sw = (unsigned short*)ws; ws += (size_t)512 * H256 * sizeof(unsigned short);
  unsigned short* W3sw = (unsigned short*)ws;

  size_t helems = (size_t)B * N * H256;
  int hblocks = (int)((helems + 255) / 256);

  // 1) z = mean + exp(logvar/2)*eps
  k_sample_z<<<(B * DIN + 255) / 256, 256, 0, stream>>>(x, eps, Wzm, bzm, Wzlv, bzlv, z, B);
  // 2) zW1 = z @ W1[0:1024] + b1
  k_zw1<<<B, 256, 0, stream>>>(z, W1, b1, zW1);
  // 3) weight swizzle (bf16 WMMA fragments)
  k_swizzle_w<<<(512 * H256 + 255) / 256, 256, 0, stream>>>(W2, W2sw, 512);
  k_swizzle_w<<<(512 * H256 + 255) / 256, 256, 0, stream>>>(W3, W3sw, 512);

  // 4) EdgeConv-1 (analytic decomposition), raw output with -inf holes
  k_fill_neginf<<<hblocks, 256, 0, stream>>>(buf0, helems);
  k_edgeconv1<<<dim3(E, B), 256, 0, stream>>>(svp, esrc, edst, W1, zW1, a1, buf0, N);

  // 5) EdgeConv-2 (WMMA), -inf fixup fused into the gather
  k_fill_neginf<<<hblocks, 256, 0, stream>>>(buf1, helems);
  k_edgeconv_wmma<<<dim3(E / EPB, B), 256, 0, stream>>>(buf0, esrc, edst, W2sw, b2, a2, buf1, N);

  // 6) EdgeConv-3 (WMMA, reuse buf0)
  k_fill_neginf<<<hblocks, 256, 0, stream>>>(buf0, helems);
  k_edgeconv_wmma<<<dim3(E / EPB, B), 256, 0, stream>>>(buf1, esrc, edst, W3sw, b3, a3, buf0, N);

  // 7) output projection 256 -> 3 (-inf fixup fused)
  k_outproj<<<(B * N + 255) / 256, 256, 0, stream>>>(buf0, Wo, bo, y, B * N);
}
